// MultiHeadSelfAttentionBlock_39831526703760
// MI455X (gfx1250) — compile-verified, matrix operand order checked by reading, not run on hardware
//
#include <hip/hip_runtime.h>
#include <hip/hip_bf16.h>

// ---------------------------------------------------------------------------
// MHSA block for MI455X (gfx1250, wave32, WMMA f32_16x16x32_f16).
// All four GEMMs + attention run through v_wmma_f32_16x16x32_f16.
// Weights pre-converted to f16 once; GEMM waves compute 16x64 strips so each
// activation load/convert feeds 4 WMMAs.
// ---------------------------------------------------------------------------

typedef __attribute__((ext_vector_type(16))) _Float16 v16h;
typedef __attribute__((ext_vector_type(8)))  _Float16 v8h;
typedef __attribute__((ext_vector_type(8)))  float    v8f;
typedef __attribute__((ext_vector_type(4)))  float    v4f;

#define BB   2
#define QSZ  2048
#define KVSZ 2048
#define EE   1024
#define HH   16
#define HD   64
#define ATT_SCALE 0.125f   // 1/sqrt(64)

__device__ __forceinline__ v8f zero8() {
    v8f z;
#pragma unroll
    for (int i = 0; i < 8; ++i) z[i] = 0.0f;
    return z;
}

__device__ __forceinline__ v8f wmma16(v16h a, v16h b, v8f c) {
    // D = A(16x32 f16) * B(32x16 f16) + C(16x16 f32)
    return __builtin_amdgcn_wmma_f32_16x16x32_f16(
        /*neg_a=*/false, a, /*neg_b=*/false, b,
        /*c_mod=*/(short)0, c, /*reuse_a=*/false, /*reuse_b=*/false);
}

// Operand loader: per-lane pattern for 16-bit A/B operands of 16x16x32 WMMA.
// Elements 0..7  = row[c .. c+7], elements 8..15 = row[c+16 .. c+23].
__device__ __forceinline__ v16h load16_f16(const _Float16* __restrict__ p, int c) {
    v8h a = *(const v8h*)(p + c);
    v8h b = *(const v8h*)(p + c + 16);
    v16h r;
#pragma unroll
    for (int i = 0; i < 8; ++i) { r[i] = a[i]; r[8 + i] = b[i]; }
    return r;
}

__device__ __forceinline__ v16h load16_f32(const float* __restrict__ p, int c) {
    v4f a0 = *(const v4f*)(p + c);
    v4f a1 = *(const v4f*)(p + c + 4);
    v4f a2 = *(const v4f*)(p + c + 16);
    v4f a3 = *(const v4f*)(p + c + 20);
    v16h r;
#pragma unroll
    for (int i = 0; i < 4; ++i) {
        r[i]      = (_Float16)a0[i];
        r[4 + i]  = (_Float16)a1[i];
        r[8 + i]  = (_Float16)a2[i];
        r[12 + i] = (_Float16)a3[i];
    }
    return r;
}

// ---------------------------------------------------------------------------
// Elementwise f32 -> f16 (used once per weight matrix). 8 elements/thread.
// ---------------------------------------------------------------------------
__global__ void __launch_bounds__(256)
cvt_f32_to_f16(const float* __restrict__ src, _Float16* __restrict__ dst) {
    const size_t i = ((size_t)blockIdx.x * blockDim.x + threadIdx.x) * 8;
    v4f a0 = *(const v4f*)(src + i);
    v4f a1 = *(const v4f*)(src + i + 4);
    v8h o;
#pragma unroll
    for (int r = 0; r < 4; ++r) { o[r] = (_Float16)a0[r]; o[4 + r] = (_Float16)a1[r]; }
    *(v8h*)(dst + i) = o;
}

// ---------------------------------------------------------------------------
// Y[s,e] = sum_k X[s,k] * W[e,k]   (Y in f16, W already f16).
// Computed as Y^T = W * X^T; one wave covers a 16(s) x 64(e) strip, so each
// converted X operand feeds 4 WMMAs.  Block = 1 wave.
// ---------------------------------------------------------------------------
__global__ void __launch_bounds__(32)
gemm_xwT_f16(const float* __restrict__ X, const _Float16* __restrict__ Wh,
             _Float16* __restrict__ Y, int Kd, int Eo) {
    const int lane = threadIdx.x & 31;
    const int lo = lane & 15, hi = lane >> 4;
    const int stile = blockIdx.x;
    const int ebase = blockIdx.y * 64;

    const float* Xrow = X + (size_t)(stile * 16 + lo) * Kd;
    const _Float16* Wr0 = Wh + (size_t)(ebase + lo) * Kd;
    const _Float16* Wr1 = Wr0 + (size_t)16 * Kd;
    const _Float16* Wr2 = Wr0 + (size_t)32 * Kd;
    const _Float16* Wr3 = Wr0 + (size_t)48 * Kd;

    v8f a0 = zero8(), a1 = zero8(), a2 = zero8(), a3 = zero8();
    for (int k = 0; k < Kd; k += 32) {
        const int c = k + hi * 8;
        const v16h bop = load16_f32(Xrow, c);      // shared across 4 WMMAs
        a0 = wmma16(load16_f16(Wr0, c), bop, a0);
        a1 = wmma16(load16_f16(Wr1, c), bop, a1);
        a2 = wmma16(load16_f16(Wr2, c), bop, a2);
        a3 = wmma16(load16_f16(Wr3, c), bop, a3);
    }
    // D(m=e_local, n=s_local): lane column = lo, rows e_local = r + 8*hi.
    _Float16* yp = Y + (size_t)(stile * 16 + lo) * Eo + ebase + hi * 8;
    v8f* acc[4] = { &a0, &a1, &a2, &a3 };
#pragma unroll
    for (int t = 0; t < 4; ++t) {
        v8h out;
#pragma unroll
        for (int r = 0; r < 8; ++r) out[r] = (_Float16)((*acc[t])[r]);
        *(v8h*)(yp + t * 16) = out;
    }
}

// Output projection: f16 activations x f16 weights -> fp32 output, 16x64 strip.
__global__ void __launch_bounds__(32)
gemm_owT_f32(const _Float16* __restrict__ Xh, const _Float16* __restrict__ Wh,
             float* __restrict__ Y, int Kd, int Eo) {
    const int lane = threadIdx.x & 31;
    const int lo = lane & 15, hi = lane >> 4;
    const int stile = blockIdx.x;
    const int ebase = blockIdx.y * 64;

    const _Float16* Xrow = Xh + (size_t)(stile * 16 + lo) * Kd;
    const _Float16* Wr0 = Wh + (size_t)(ebase + lo) * Kd;
    const _Float16* Wr1 = Wr0 + (size_t)16 * Kd;
    const _Float16* Wr2 = Wr0 + (size_t)32 * Kd;
    const _Float16* Wr3 = Wr0 + (size_t)48 * Kd;

    v8f a0 = zero8(), a1 = zero8(), a2 = zero8(), a3 = zero8();
    for (int k = 0; k < Kd; k += 32) {
        const int c = k + hi * 8;
        const v16h bop = load16_f16(Xrow, c);
        a0 = wmma16(load16_f16(Wr0, c), bop, a0);
        a1 = wmma16(load16_f16(Wr1, c), bop, a1);
        a2 = wmma16(load16_f16(Wr2, c), bop, a2);
        a3 = wmma16(load16_f16(Wr3, c), bop, a3);
    }
    float* yp = Y + (size_t)(stile * 16 + lo) * Eo + ebase + hi * 8;
    v8f* acc[4] = { &a0, &a1, &a2, &a3 };
#pragma unroll
    for (int t = 0; t < 4; ++t) {
        v4f o0, o1;
#pragma unroll
        for (int r = 0; r < 4; ++r) { o0[r] = (*acc[t])[r]; o1[r] = (*acc[t])[4 + r]; }
        *(v4f*)(yp + t * 16)     = o0;
        *(v4f*)(yp + t * 16 + 4) = o1;
    }
}

// ---------------------------------------------------------------------------
// Flash-style attention, one wave per (batch, head, 16-row q-tile).
// S^T tiles = K * Q^T  (softmax axis lands in the VGPR/M dimension),
// O^T tiles = V^T * P^T (P^T B-operand == post-softmax registers, no shuffle).
// ---------------------------------------------------------------------------
__global__ void __launch_bounds__(32)
flash_attn(const _Float16* __restrict__ Qb, const _Float16* __restrict__ Kb,
           const _Float16* __restrict__ Vb, const int* __restrict__ mask,
           _Float16* __restrict__ O) {
    __shared__ _Float16 Vt[HD][40];   // V^T tile, 80B row stride (16B aligned chunks)

    const int lane = threadIdx.x & 31;
    const int lo = lane & 15, hi = lane >> 4;
    const int qtiles = QSZ / 16;
    const int qt = blockIdx.x % qtiles;
    const int h  = (blockIdx.x / qtiles) % HH;
    const int b  = blockIdx.x / (qtiles * HH);
    const int q  = qt * 16 + lo;

    // Q operand (B-matrix): lane n holds Q[q=n, hd pattern] -> row-major load.
    const _Float16* Qrow = Qb + ((size_t)b * QSZ + q) * EE + h * HD;
    const v16h qop0 = load16_f16(Qrow, hi * 8);         // hd 0..31
    const v16h qop1 = load16_f16(Qrow, 32 + hi * 8);    // hd 32..63

    v8f o0 = zero8(), o1 = zero8(), o2 = zero8(), o3 = zero8();
    float mrun = -1e30f, lrun = 0.0f;
    const int* maskrow = mask + (size_t)q * KVSZ;

    for (int kv0 = 0; kv0 < KVSZ; kv0 += 32) {
        // ---- stage V^T tile (32 kv x 64 hd) into LDS, transposed ----
        const _Float16* Vrow = Vb + ((size_t)b * KVSZ + kv0 + lane) * EE + h * HD;
#pragma unroll
        for (int j = 0; j < 8; ++j) {
            v8h vv = *(const v8h*)(Vrow + j * 8);
#pragma unroll
            for (int t = 0; t < 8; ++t) Vt[j * 8 + t][lane] = vv[t];
        }
        __syncthreads();

        // ---- S^T tiles: s0 = kv[0..15] x q[0..15], s1 = kv[16..31] ----
        const _Float16* Krow0 = Kb + ((size_t)b * KVSZ + kv0 + lo) * EE + h * HD;
        const _Float16* Krow1 = Krow0 + (size_t)16 * EE;
        v8f s0 = zero8(), s1 = zero8();
        s0 = wmma16(load16_f16(Krow0, hi * 8),      qop0, s0);
        s0 = wmma16(load16_f16(Krow0, 32 + hi * 8), qop1, s0);
        s1 = wmma16(load16_f16(Krow1, hi * 8),      qop0, s1);
        s1 = wmma16(load16_f16(Krow1, 32 + hi * 8), qop1, s1);

        // ---- mask (fill 1e-20, matching reference) + scale + tile max ----
        float tmax = -1e30f;
#pragma unroll
        for (int r = 0; r < 8; ++r) {
            const int m0 = maskrow[kv0 + hi * 8 + r];
            const int m1 = maskrow[kv0 + 16 + hi * 8 + r];
            s0[r] = m0 ? s0[r] * ATT_SCALE : 1e-20f;
            s1[r] = m1 ? s1[r] * ATT_SCALE : 1e-20f;
            tmax = fmaxf(tmax, fmaxf(s0[r], s1[r]));
        }
        tmax = fmaxf(tmax, __shfl_xor(tmax, 16, 32));   // merge half-lane rows

        // ---- online softmax update ----
        const float mnew  = fmaxf(mrun, tmax);
        const float alpha = __expf(mrun - mnew);
        lrun *= alpha;
#pragma unroll
        for (int r = 0; r < 8; ++r) {
            o0[r] *= alpha; o1[r] *= alpha; o2[r] *= alpha; o3[r] *= alpha;
        }
        v16h pb;          // P^T B-operand: exactly the per-lane kv pattern
        float ls = 0.0f;
#pragma unroll
        for (int r = 0; r < 8; ++r) {
            const float p0 = __expf(s0[r] - mnew);
            const float p1 = __expf(s1[r] - mnew);
            ls += p0 + p1;
            pb[r]     = (_Float16)p0;
            pb[8 + r] = (_Float16)p1;
        }
        lrun += ls;
        mrun = mnew;

        // ---- O^T(hd x q) += V^T(hd x kv) * P^T(kv x q), 4 hd-tiles ----
#pragma unroll
        for (int t = 0; t < 4; ++t) {
            const _Float16* vr = &Vt[t * 16 + lo][0];
            v8h x0 = *(const v8h*)(vr + hi * 8);
            v8h x1 = *(const v8h*)(vr + 16 + hi * 8);
            v16h va;
#pragma unroll
            for (int i = 0; i < 8; ++i) { va[i] = x0[i]; va[8 + i] = x1[i]; }
            v8f& ot = (t == 0) ? o0 : (t == 1) ? o1 : (t == 2) ? o2 : o3;
            ot = wmma16(va, pb, ot);
        }
        __syncthreads();
    }

    // ---- finalize: combine half-lane partial sums, normalize, store f16 ----
    const float ltot = lrun + __shfl_xor(lrun, 16, 32);
    const float inv  = 1.0f / ltot;
    _Float16* orow = O + ((size_t)b * QSZ + q) * EE + h * HD;
    v8f* oacc[4] = { &o0, &o1, &o2, &o3 };
#pragma unroll
    for (int t = 0; t < 4; ++t) {
        v8h w;
#pragma unroll
        for (int r = 0; r < 8; ++r) w[r] = (_Float16)((*oacc[t])[r] * inv);
        *(v8h*)(orow + t * 16 + hi * 8) = w;
    }
}

// ---------------------------------------------------------------------------
// Host-side launch. Input order: v_in, k_in, q_in, mask, Wv, Wk, Wq, Wout.
// Workspace: Qf | Kf | Vf | Of (8.39 MB each) | Wqf | Wkf | Wvf | Woutf (2 MB
// each)  -> ~40 MB total.
// ---------------------------------------------------------------------------
extern "C" void kernel_launch(void* const* d_in, const int* in_sizes, int n_in,
                              void* d_out, int out_size, void* d_ws, size_t ws_size,
                              hipStream_t stream) {
    (void)in_sizes; (void)n_in; (void)out_size; (void)ws_size;
    const float* v_in = (const float*)d_in[0];
    const float* k_in = (const float*)d_in[1];
    const float* q_in = (const float*)d_in[2];
    const int*   mask = (const int*)d_in[3];
    const float* Wv   = (const float*)d_in[4];
    const float* Wk   = (const float*)d_in[5];
    const float* Wq   = (const float*)d_in[6];
    const float* Wout = (const float*)d_in[7];
    float* out = (float*)d_out;

    const size_t N  = (size_t)BB * QSZ * EE;   // 4.19M elements, activations
    const size_t NW = (size_t)EE * EE;         // 1.05M elements, weights
    _Float16* Qf    = (_Float16*)d_ws;
    _Float16* Kf    = Qf + N;
    _Float16* Vf    = Kf + N;
    _Float16* Of    = Vf + N;
    _Float16* Wqf   = Of + N;
    _Float16* Wkf   = Wqf + NW;
    _Float16* Wvf   = Wkf + NW;
    _Float16* Woutf = Wvf + NW;

    // 1) weights f32 -> f16 (once)
    const int cvt_blocks = (int)(NW / (256 * 8));          // 512 blocks
    cvt_f32_to_f16<<<cvt_blocks, 256, 0, stream>>>(Wq,   Wqf);
    cvt_f32_to_f16<<<cvt_blocks, 256, 0, stream>>>(Wk,   Wkf);
    cvt_f32_to_f16<<<cvt_blocks, 256, 0, stream>>>(Wv,   Wvf);
    cvt_f32_to_f16<<<cvt_blocks, 256, 0, stream>>>(Wout, Woutf);

    // 2) Q/K/V projections (16x64 strips per wave)
    dim3 grid(BB * QSZ / 16, EE / 64);                     // 256 x 16
    gemm_xwT_f16<<<grid, 32, 0, stream>>>(q_in, Wqf, Qf, EE, EE);
    gemm_xwT_f16<<<grid, 32, 0, stream>>>(k_in, Wkf, Kf, EE, EE);
    gemm_xwT_f16<<<grid, 32, 0, stream>>>(v_in, Wvf, Vf, EE, EE);

    // 3) attention
    flash_attn<<<BB * HH * (QSZ / 16), 32, 0, stream>>>(Qf, Kf, Vf, mask, Of);

    // 4) output projection -> fp32 d_out
    gemm_owT_f32<<<grid, 32, 0, stream>>>(Of, Woutf, out, EE, EE);
}
